// V2CGNN_73650099192329
// MI455X (gfx1250) — matrix operand order repaired and checked by reading, not compiled, not run on hardware
//
#include <hip/hip_runtime.h>
#include <hip/hip_bf16.h>

// ---------------------------------------------------------------------------
// CDNA5 (gfx1250, wave32) implementation.
//
// CSR build: histogram dst -> exclusive scan -> fill per-dst edge lists.
// Phase A: var/negvar nodes GATHER pos_state over their edge lists (16 lanes
//          per node, fp32 register accumulation; no float atomics).
// Phase B: per-16-node fused GEMM chain via v_wmma_f32_16x16x32_bf16:
//          msg = agg16 @ Wr + deg*br ; h=[msg|x] ; MLP 67->128->128->64
// Phase C: clause nodes GATHER literal states (wave32 per node).
// Phase D: msg = agg64 @ Wc + deg*bc ; MLP -> clause states
// Phase E: predictor MLP 64->128->128->2 over all N nodes -> d_out (f32)
// ---------------------------------------------------------------------------

typedef __attribute__((ext_vector_type(16))) __bf16 v16bf;
typedef __attribute__((ext_vector_type(8)))  __bf16 v8bf;
typedef __attribute__((ext_vector_type(8)))  float  v8f;

#define WAVES 8        // waves per block (256 threads) in WMMA kernels
#define SCAN_CHUNK 1024

// Within-wave LDS ordering: DS ops are in-order per wave and lanes are in
// lockstep, so a dscnt wait is sufficient for cross-lane LDS RAW.
__device__ __forceinline__ void lds_wave_fence() {
  asm volatile("s_wait_dscnt 0" ::: "memory");
}

// A-operand (16x32 bf16, MxK): lane holds row M=lane%16; half=lane/16 selects
// K = {k0+16h .. k0+16h+15} in v[0..15] (contiguous, matches B).
__device__ __forceinline__ v16bf load_a16(const __bf16* row, int k0, int half) {
  v8bf lo = *(const v8bf*)(row + k0 + 16 * half);
  v8bf hi = *(const v8bf*)(row + k0 + 16 * half + 8);
  return __builtin_shufflevector(lo, hi, 0, 1, 2, 3, 4, 5, 6, 7,
                                 8, 9, 10, 11, 12, 13, 14, 15);
}

__device__ __forceinline__ v8bf cvt8(const float* p) {
  v8bf r;
#pragma unroll
  for (int i = 0; i < 8; i++) r[i] = (__bf16)p[i];
  return r;
}

__device__ __forceinline__ v8bf zero8() {
  v8bf r;
#pragma unroll
  for (int i = 0; i < 8; i++) r[i] = (__bf16)0.0f;
  return r;
}

// One 16-node tile x one dense layer:
//   out[16][NOUT] = act( A[16][KPAD] @ WT^T + bias )   (WT is [NOUT][KPAD] bf16)
// DEG -> bias term is deg[row]*bias[col] (DeepSet per-edge bias).
template <int KPAD, int NOUT, int OSTR, int ACT, bool DEG>
__device__ __forceinline__ void gemm16(const __bf16* __restrict__ A,
                                       const __bf16* __restrict__ WT,
                                       const float* __restrict__ bias,
                                       const float* __restrict__ deg,
                                       __bf16* __restrict__ outB, int lane) {
  constexpr int NK = KPAD / 32;
  const int m = lane & 15, half = lane >> 4;
  const __bf16* arow = A + (size_t)m * KPAD;

  v16bf af[NK];
#pragma unroll
  for (int k = 0; k < NK; k++) af[k] = load_a16(arow, 32 * k, half);

  float dr[8];
#pragma unroll
  for (int r = 0; r < 8; r++) dr[r] = DEG ? deg[r + 8 * half] : 0.f;

#pragma unroll
  for (int n0 = 0; n0 < NOUT; n0 += 16) {
    const int n = n0 + m;
    v8f acc = {0.f, 0.f, 0.f, 0.f, 0.f, 0.f, 0.f, 0.f};
    const __bf16* wrow = WT + (size_t)n * KPAD + 16 * half;
#pragma unroll
    for (int k = 0; k < NK; k++) {
      v16bf b = *(const v16bf*)(wrow + 32 * k);
      acc = __builtin_amdgcn_wmma_f32_16x16x32_bf16(false, af[k], false, b,
                                                    (short)0, acc, false, false);
    }
    const float bv = bias[n];
#pragma unroll
    for (int r = 0; r < 8; r++) {
      float v = DEG ? (acc[r] + dr[r] * bv) : (acc[r] + bv);
      if (ACT) v = fmaxf(v, 0.f);
      outB[(size_t)(r + 8 * half) * OSTR + n] = (__bf16)v;
    }
  }
  lds_wave_fence();
}

// Stage weight W[K][N] (f32, row-major) into LDS transposed [NPAD][KPAD] bf16.
__device__ __forceinline__ void stage_wt(const float* __restrict__ W,
                                         __bf16* __restrict__ WT, int K, int N,
                                         int KPAD, int NPAD, int tid, int nth) {
  const int tot = NPAD * KPAD;
  for (int i = tid; i < tot; i += nth) {
    const int n = i / KPAD, k = i - n * KPAD;
    float v = (k < K && n < N) ? W[(size_t)k * N + n] : 0.f;
    WT[i] = (__bf16)v;
  }
}

__device__ __forceinline__ void stage_b(const float* __restrict__ b,
                                        float* __restrict__ lb, int Nv, int NPAD,
                                        int tid, int nth) {
  for (int i = tid; i < NPAD; i += nth) lb[i] = (i < Nv) ? b[i] : 0.f;
}

// ---------------------------------------------------------------------------
// CSR build over destinations.
// ---------------------------------------------------------------------------
__global__ __launch_bounds__(256) void csr_count(const int* __restrict__ ei,
                                                 int* __restrict__ cnt, int E) {
  const int e = blockIdx.x * blockDim.x + threadIdx.x;
  if (e >= E) return;
  atomicAdd(&cnt[ei[(size_t)E + e]], 1);
}

__global__ __launch_bounds__(256) void scan_block_sums(
    const int* __restrict__ cnt, int* __restrict__ part, int N) {
  __shared__ int sm[256];
  const int b = blockIdx.x, t = threadIdx.x;
  const int base = b * SCAN_CHUNK;
  int sum = 0;
  for (int i = t; i < SCAN_CHUNK; i += 256) {
    const int idx = base + i;
    sum += (idx < N) ? cnt[idx] : 0;
  }
  sm[t] = sum;
  __syncthreads();
  for (int s = 128; s > 0; s >>= 1) {
    if (t < s) sm[t] += sm[t + s];
    __syncthreads();
  }
  if (t == 0) part[b] = sm[0];
}

// Single-block exclusive scan of nb (<=1024) partial sums.
__global__ __launch_bounds__(1024) void scan_partials(int* __restrict__ part,
                                                      int nb) {
  __shared__ int sm[1024];
  const int t = threadIdx.x;
  const int orig = (t < nb) ? part[t] : 0;
  sm[t] = orig;
  __syncthreads();
  for (int o = 1; o < 1024; o <<= 1) {
    const int v = (t >= o) ? sm[t - o] : 0;
    __syncthreads();
    sm[t] += v;
    __syncthreads();
  }
  if (t < nb) part[t] = sm[t] - orig;  // exclusive
}

__global__ __launch_bounds__(256) void scan_final(
    const int* __restrict__ cnt, const int* __restrict__ part,
    int* __restrict__ ofs, int* __restrict__ fill, int N) {
  __shared__ int sm[256];
  const int b = blockIdx.x, t = threadIdx.x;
  const int i0 = b * SCAN_CHUNK + t * 4;
  int c[4];
#pragma unroll
  for (int q = 0; q < 4; q++) c[q] = (i0 + q < N) ? cnt[i0 + q] : 0;
  const int tsum = c[0] + c[1] + c[2] + c[3];
  sm[t] = tsum;
  __syncthreads();
  for (int o = 1; o < 256; o <<= 1) {
    const int v = (t >= o) ? sm[t - o] : 0;
    __syncthreads();
    sm[t] += v;
    __syncthreads();
  }
  int run = sm[t] - tsum + part[b];  // exclusive offset for this thread
#pragma unroll
  for (int q = 0; q < 4; q++) {
    if (i0 + q < N) {
      ofs[i0 + q] = run;
      fill[i0 + q] = run;
      run += c[q];
    }
  }
}

__global__ __launch_bounds__(256) void csr_fill(const int* __restrict__ ei,
                                                int* __restrict__ fill,
                                                int* __restrict__ elist, int E) {
  const int e = blockIdx.x * blockDim.x + threadIdx.x;
  if (e >= E) return;
  const int d = ei[(size_t)E + e];
  const int p = atomicAdd(&fill[d], 1);
  elist[p] = ei[e];
}

// ---------------------------------------------------------------------------
// Phase A (gather): 16 lanes per var/negvar node; lane k accumulates pos
// element k over the node's incoming edge list. No atomics.
// ---------------------------------------------------------------------------
__global__ __launch_bounds__(256) void var_gather(
    const int* __restrict__ ofs, const int* __restrict__ cnt,
    const int* __restrict__ elist, const float* __restrict__ pos,
    float* __restrict__ agg16, float* __restrict__ deg16, int nlit, int nvar) {
  const int t = blockIdx.x * blockDim.x + threadIdx.x;
  const int d = t >> 4, k = t & 15;
  if (d >= nlit) return;
  const int base = ofs[d], n = cnt[d];
  float acc = 0.f;
  for (int j = 0; j < n; j++) {
    int s = elist[base + j];        // uniform across the 16-lane group
    if (s < 2 * nvar) {             // pos_state row nonzero
      float sg = 1.f;
      if (s >= nvar) { sg = -1.f; s -= nvar; }
      acc += sg * pos[(size_t)s * 16 + k];
    }
  }
  agg16[(size_t)d * 16 + k] = acc;
  if (k == 0) deg16[d] = (float)n;  // bias counts every masked edge
}

// ---------------------------------------------------------------------------
// Phase C (gather): one wave32 per clause node; lane l accumulates state
// elements l and l+32. node_state rows >= 2*nvar are zero at this point.
// ---------------------------------------------------------------------------
__global__ __launch_bounds__(256) void cla_gather(
    const int* __restrict__ ofs, const int* __restrict__ cnt,
    const int* __restrict__ elist, const __bf16* __restrict__ nstate,
    float* __restrict__ agg64, float* __restrict__ deg64, int ncla, int nvar) {
  const int wv = (blockIdx.x * blockDim.x + threadIdx.x) >> 5;
  const int lane = threadIdx.x & 31;
  if (wv >= ncla) return;
  const int d = 2 * nvar + wv;
  const int base = ofs[d], n = cnt[d];
  float a0 = 0.f, a1 = 0.f;
  for (int j = 0; j < n; j++) {
    const int s = elist[base + j];  // uniform across the wave
    if (s < 2 * nvar) {
      const __bf16* h = nstate + (size_t)s * 64;
      a0 += (float)h[lane];
      a1 += (float)h[lane + 32];
    }
  }
  agg64[(size_t)wv * 64 + lane] = a0;
  agg64[(size_t)wv * 64 + lane + 32] = a1;
  if (lane == 0) deg64[wv] = (float)n;
}

// ---------------------------------------------------------------------------
// Phase B: var/negvar nodes (nNodes = 2*nvar, multiple of 16).
// ---------------------------------------------------------------------------
__global__ __launch_bounds__(256) void var_update_kernel(
    const float* __restrict__ agg16, const float* __restrict__ deg16,
    const float* __restrict__ x, const float* __restrict__ Wr,
    const float* __restrict__ br, const float* __restrict__ uW1,
    const float* __restrict__ ub1, const float* __restrict__ uW2,
    const float* __restrict__ ub2, const float* __restrict__ uW3,
    const float* __restrict__ ub3, __bf16* __restrict__ nstate, int nNodes) {
  __shared__ __align__(32) __bf16 sWr[64 * 32];
  __shared__ __align__(32) __bf16 sW1[128 * 96];
  __shared__ __align__(32) __bf16 sW2[128 * 128];
  __shared__ __align__(32) __bf16 sW3[64 * 128];
  __shared__ float sbr[64], sb1[128], sb2[128], sb3[64];
  __shared__ __align__(32) __bf16 actA[WAVES][16 * 128];
  __shared__ __align__(32) __bf16 actB[WAVES][16 * 128];
  __shared__ float sdeg[WAVES][16];

  const int tid = threadIdx.x, nth = blockDim.x;
  stage_wt(Wr, sWr, 16, 64, 32, 64, tid, nth);
  stage_wt(uW1, sW1, 67, 128, 96, 128, tid, nth);
  stage_wt(uW2, sW2, 128, 128, 128, 128, tid, nth);
  stage_wt(uW3, sW3, 128, 64, 128, 64, tid, nth);
  stage_b(br, sbr, 64, 64, tid, nth);
  stage_b(ub1, sb1, 128, 128, tid, nth);
  stage_b(ub2, sb2, 128, 128, tid, nth);
  stage_b(ub3, sb3, 64, 64, tid, nth);
  __syncthreads();

  const int w = tid >> 5, lane = tid & 31;
  const int tile = blockIdx.x * WAVES + w;
  if (tile * 16 >= nNodes) return;
  const int g0 = tile * 16;
  __bf16* A = actA[w];
  __bf16* B = actB[w];
  float* dg = sdeg[w];

  if (lane < 16) {
    const int g = g0 + lane;
    const float* p = agg16 + (size_t)g * 16;
    v8bf* brow = (v8bf*)(B + lane * 32);
    brow[0] = cvt8(p);
    brow[1] = cvt8(p + 8);
    brow[2] = zero8();
    brow[3] = zero8();
    dg[lane] = deg16[g];
  }
  lds_wave_fence();

  // msg = agg16 @ Wr + deg*br  -> A[:, 0:64] (row stride 96 for concat)
  gemm16<32, 64, 96, 0, true>(B, sWr, sbr, dg, A, lane);

  if (lane < 16) {  // concat node features, zero-pad K 67..95
    const int g = g0 + lane;
    float xf[8] = {x[(size_t)g * 3 + 0], x[(size_t)g * 3 + 1],
                   x[(size_t)g * 3 + 2], 0.f, 0.f, 0.f, 0.f, 0.f};
    v8bf* arow = (v8bf*)(A + lane * 96 + 64);
    arow[0] = cvt8(xf);
    arow[1] = zero8();
    arow[2] = zero8();
    arow[3] = zero8();
  }
  lds_wave_fence();

  gemm16<96, 128, 128, 1, false>(A, sW1, sb1, nullptr, B, lane);
  gemm16<128, 128, 128, 1, false>(B, sW2, sb2, nullptr, A, lane);
  gemm16<128, 64, 64, 0, false>(A, sW3, sb3, nullptr,
                                nstate + (size_t)g0 * 64, lane);
}

// ---------------------------------------------------------------------------
// Phase D: clause nodes (nNodes = N - 2*nvar, multiple of 16).
// agg64 / x / nstate pointers are pre-offset to the clause partition.
// ---------------------------------------------------------------------------
__global__ __launch_bounds__(256) void cla_update_kernel(
    const float* __restrict__ agg64, const float* __restrict__ deg64,
    const float* __restrict__ x, const float* __restrict__ Wc,
    const float* __restrict__ bc, const float* __restrict__ uW1,
    const float* __restrict__ ub1, const float* __restrict__ uW2,
    const float* __restrict__ ub2, const float* __restrict__ uW3,
    const float* __restrict__ ub3, __bf16* __restrict__ nstate, int nNodes) {
  __shared__ __align__(32) __bf16 sWc[64 * 64];
  __shared__ __align__(32) __bf16 sW1[128 * 96];
  __shared__ __align__(32) __bf16 sW2[128 * 128];
  __shared__ __align__(32) __bf16 sW3[64 * 128];
  __shared__ float sbc[64], sb1[128], sb2[128], sb3[64];
  __shared__ __align__(32) __bf16 actA[WAVES][16 * 128];
  __shared__ __align__(32) __bf16 actB[WAVES][16 * 128];
  __shared__ float sdeg[WAVES][16];

  const int tid = threadIdx.x, nth = blockDim.x;
  stage_wt(Wc, sWc, 64, 64, 64, 64, tid, nth);
  stage_wt(uW1, sW1, 67, 128, 96, 128, tid, nth);
  stage_wt(uW2, sW2, 128, 128, 128, 128, tid, nth);
  stage_wt(uW3, sW3, 128, 64, 128, 64, tid, nth);
  stage_b(bc, sbc, 64, 64, tid, nth);
  stage_b(ub1, sb1, 128, 128, tid, nth);
  stage_b(ub2, sb2, 128, 128, tid, nth);
  stage_b(ub3, sb3, 64, 64, tid, nth);
  __syncthreads();

  const int w = tid >> 5, lane = tid & 31;
  const int tile = blockIdx.x * WAVES + w;
  if (tile * 16 >= nNodes) return;
  const int g0 = tile * 16;
  __bf16* A = actA[w];
  __bf16* B = actB[w];
  float* dg = sdeg[w];

  if (lane < 16) {
    const int g = g0 + lane;
    const float* p = agg64 + (size_t)g * 64;
    v8bf* brow = (v8bf*)(B + lane * 64);
#pragma unroll
    for (int i = 0; i < 8; i++) brow[i] = cvt8(p + 8 * i);
    dg[lane] = deg64[g];
  }
  lds_wave_fence();

  gemm16<64, 64, 96, 0, true>(B, sWc, sbc, dg, A, lane);  // msg -> A[:, 0:64]

  if (lane < 16) {
    const int g = g0 + lane;
    float xf[8] = {x[(size_t)g * 3 + 0], x[(size_t)g * 3 + 1],
                   x[(size_t)g * 3 + 2], 0.f, 0.f, 0.f, 0.f, 0.f};
    v8bf* arow = (v8bf*)(A + lane * 96 + 64);
    arow[0] = cvt8(xf);
    arow[1] = zero8();
    arow[2] = zero8();
    arow[3] = zero8();
  }
  lds_wave_fence();

  gemm16<96, 128, 128, 1, false>(A, sW1, sb1, nullptr, B, lane);
  gemm16<128, 128, 128, 1, false>(B, sW2, sb2, nullptr, A, lane);
  gemm16<128, 64, 64, 0, false>(A, sW3, sb3, nullptr,
                                nstate + (size_t)g0 * 64, lane);
}

// ---------------------------------------------------------------------------
// Phase E: predictor over all N nodes. Layer-1 A operand is read straight
// from global bf16 node_state (contiguous 16B runs per lane -> no staging).
// ---------------------------------------------------------------------------
__global__ __launch_bounds__(256) void predict_kernel(
    const __bf16* __restrict__ nstate, const float* __restrict__ pW1,
    const float* __restrict__ pb1, const float* __restrict__ pW2,
    const float* __restrict__ pb2, const float* __restrict__ pW3,
    const float* __restrict__ pb3, float* __restrict__ out, int N) {
  __shared__ __align__(32) __bf16 sW1[128 * 64];
  __shared__ __align__(32) __bf16 sW2[128 * 128];
  __shared__ __align__(32) __bf16 sW3[16 * 128];
  __shared__ float sb1[128], sb2[128], sb3[16];
  __shared__ __align__(32) __bf16 actA[WAVES][16 * 128];
  __shared__ __align__(32) __bf16 actB[WAVES][16 * 128];

  const int tid = threadIdx.x, nth = blockDim.x;
  stage_wt(pW1, sW1, 64, 128, 64, 128, tid, nth);
  stage_wt(pW2, sW2, 128, 128, 128, 128, tid, nth);
  stage_wt(pW3, sW3, 128, 2, 128, 16, tid, nth);
  stage_b(pb1, sb1, 128, 128, tid, nth);
  stage_b(pb2, sb2, 128, 128, tid, nth);
  stage_b(pb3, sb3, 2, 16, tid, nth);
  __syncthreads();

  const int w = tid >> 5, lane = tid & 31;
  const int tile = blockIdx.x * WAVES + w;
  if (tile * 16 >= N) return;
  const int g0 = tile * 16;
  __bf16* A = actA[w];
  __bf16* B = actB[w];

  gemm16<64, 128, 128, 1, false>(nstate + (size_t)g0 * 64, sW1, sb1, nullptr,
                                 B, lane);
  gemm16<128, 128, 128, 1, false>(B, sW2, sb2, nullptr, A, lane);

  // Final layer: N padded 2->16, only columns 0..1 written (f32 output).
  {
    const int m = lane & 15, half = lane >> 4;
    const __bf16* arow = A + (size_t)m * 128;
    const __bf16* wrow = sW3 + (size_t)m * 128 + 16 * half;
    v8f acc = {0.f, 0.f, 0.f, 0.f, 0.f, 0.f, 0.f, 0.f};
#pragma unroll
    for (int k0 = 0; k0 < 128; k0 += 32) {
      v16bf a = load_a16(arow, k0, half);
      v16bf b = *(const v16bf*)(wrow + k0);
      acc = __builtin_amdgcn_wmma_f32_16x16x32_bf16(false, a, false, b,
                                                    (short)0, acc, false, false);
    }
    if (m < 2) {
      const float bv = sb3[m];
#pragma unroll
      for (int r = 0; r < 8; r++)
        out[(size_t)(g0 + r + 8 * half) * 2 + m] = acc[r] + bv;
    }
  }
}

// ---------------------------------------------------------------------------
extern "C" void kernel_launch(void* const* d_in, const int* in_sizes, int n_in,
                              void* d_out, int out_size, void* d_ws,
                              size_t ws_size, hipStream_t stream) {
  const float* x   = (const float*)d_in[0];
  const int*   ei  = (const int*)d_in[1];
  const float* pos = (const float*)d_in[3];
  const float* Wr  = (const float*)d_in[4];
  const float* br  = (const float*)d_in[5];
  const float* Wc  = (const float*)d_in[6];
  const float* bc  = (const float*)d_in[7];
  const float* uW1 = (const float*)d_in[8];
  const float* ub1 = (const float*)d_in[9];
  const float* uW2 = (const float*)d_in[10];
  const float* ub2 = (const float*)d_in[11];
  const float* uW3 = (const float*)d_in[12];
  const float* ub3 = (const float*)d_in[13];
  const float* pW1 = (const float*)d_in[14];
  const float* pb1 = (const float*)d_in[15];
  const float* pW2 = (const float*)d_in[16];
  const float* pb2 = (const float*)d_in[17];
  const float* pW3 = (const float*)d_in[18];
  const float* pb3 = (const float*)d_in[19];

  const int N    = in_sizes[2];
  const int E    = in_sizes[1] / 2;
  const int NVR  = in_sizes[3] / 16;  // nvar (pos_random is [nvar,16])
  const int NLIT = 2 * NVR;
  const int NCLA = N - NLIT;
  const int NB   = (N + SCAN_CHUNK - 1) / SCAN_CHUNK;  // scan blocks (<=1024)

  // Workspace carve-out (256B aligned slices).
  char* ws = (char*)d_ws;
  size_t off = 0;
  auto carve = [&](size_t bytes) {
    size_t p = off;
    off = (off + bytes + 255) & ~(size_t)255;
    return p;
  };
  float*  agg16  = (float*)(ws + carve((size_t)NLIT * 16 * 4));
  float*  deg16  = (float*)(ws + carve((size_t)NLIT * 4));
  float*  agg64  = (float*)(ws + carve((size_t)NCLA * 64 * 4));
  float*  deg64  = (float*)(ws + carve((size_t)NCLA * 4));
  __bf16* nstate = (__bf16*)(ws + carve((size_t)N * 64 * 2));
  int*    cnt    = (int*)(ws + carve((size_t)N * 4));
  int*    ofs    = (int*)(ws + carve((size_t)N * 4));
  int*    fill   = (int*)(ws + carve((size_t)N * 4));
  int*    part   = (int*)(ws + carve((size_t)1024 * 4));
  int*    elist  = (int*)(ws + carve((size_t)E * 4));

  // ---- CSR build (only cnt needs zeroing) ----
  (void)hipMemsetAsync(cnt, 0, (size_t)N * 4, stream);
  csr_count<<<(E + 255) / 256, 256, 0, stream>>>(ei, cnt, E);
  scan_block_sums<<<NB, 256, 0, stream>>>(cnt, part, N);
  scan_partials<<<1, 1024, 0, stream>>>(part, NB);
  scan_final<<<NB, 256, 0, stream>>>(cnt, part, ofs, fill, N);
  csr_fill<<<(E + 255) / 256, 256, 0, stream>>>(ei, fill, elist, E);

  // A: gather pos features into var/negvar aggregates (16 lanes per node)
  var_gather<<<((size_t)NLIT * 16 + 255) / 256, 256, 0, stream>>>(
      ofs, cnt, elist, pos, agg16, deg16, NLIT, NVR);
  // B: var/negvar DeepSet GEMM + update MLP (WMMA)
  {
    const int tiles = NLIT / 16;
    var_update_kernel<<<(tiles + WAVES - 1) / WAVES, 32 * WAVES, 0, stream>>>(
        agg16, deg16, x, Wr, br, uW1, ub1, uW2, ub2, uW3, ub3, nstate, NLIT);
  }
  // C: gather literal states into clause aggregates (wave32 per node)
  cla_gather<<<((size_t)NCLA * 32 + 255) / 256, 256, 0, stream>>>(
      ofs, cnt, elist, nstate, agg64, deg64, NCLA, NVR);
  // D: clause DeepSet GEMM + update MLP (WMMA)
  {
    const int tiles = NCLA / 16;
    cla_update_kernel<<<(tiles + WAVES - 1) / WAVES, 32 * WAVES, 0, stream>>>(
        agg64, deg64, x + (size_t)NLIT * 3, Wc, bc, uW1, ub1, uW2, ub2, uW3,
        ub3, nstate + (size_t)NLIT * 64, NCLA);
  }
  // E: predictor MLP over all nodes (WMMA) -> f32 output [N,2]
  {
    const int tiles = N / 16;
    predict_kernel<<<(tiles + WAVES - 1) / WAVES, 32 * WAVES, 0, stream>>>(
        nstate, pW1, pb1, pW2, pb2, pW3, pb3, (float*)d_out, N);
  }
}
